// MoeLayer_45621142618738
// MI455X (gfx1250) — compile-verified, matrix operand order checked by reading, not run on hardware
//
#include <hip/hip_runtime.h>
#include <hip/hip_bf16.h>
#include <math.h>

// ---------------- problem constants (from reference) ----------------
#define NB      2
#define NS      2048
#define NTOK    (NB * NS)        // 4096 tokens
#define DIM     1024
#define HIDDEN  2816
#define NE      8
#define NK      2                // top-k
#define TILE_M  128              // slot-tile rows
#define TILE_N  64               // output cols per block
#define TILE_K  32               // bf16 WMMA K
#define SLOT_CAP (NTOK * NK + NE * TILE_M)   // 9216: 128-aligned expert segments

typedef __attribute__((ext_vector_type(16))) __bf16 bf16x16;
typedef __attribute__((ext_vector_type(8)))  float  f32x8;
union Frag16 { bf16x16 v; unsigned int u[8]; };

__device__ __forceinline__ unsigned short f32_to_bf16(float f) {
    union { float f; unsigned int u; } v; v.f = f;
    unsigned int r = v.u + 0x7FFFu + ((v.u >> 16) & 1u);   // round-nearest-even
    return (unsigned short)(r >> 16);
}
// VGPR i -> K-pair index within a 16-pair (32 bf16) row, per ISA 16-bit A/B layout
__device__ __forceinline__ int pair_idx(int i, int half) {
    return (i & 3) + ((i >> 2) << 3) + half * 4;
}
__device__ __forceinline__ float fast_sigmoid(float x) {
    return __builtin_amdgcn_rcpf(1.0f + __expf(-x));       // v_exp + v_rcp
}

// ---------------- init: zero out, slot tables, counts ----------------
__global__ void init_kernel(float* out, int* slot_tok, float* slot_w, int* counts) {
    int i = blockIdx.x * blockDim.x + threadIdx.x;
    if (i < NTOK * DIM) out[i] = 0.0f;
    if (i < SLOT_CAP) { slot_tok[i] = 0; slot_w[i] = 0.0f; }
    if (i < NE) counts[i] = 0;
}

// ---------------- fp32 -> bf16 conversion ----------------
__global__ void cvt_bf16_kernel(const float* __restrict__ s, unsigned short* __restrict__ d, int n) {
    int i = blockIdx.x * blockDim.x + threadIdx.x;
    if (i < n) d[i] = f32_to_bf16(s[i]);
}

// ---------------- gating: one wave32 per token ----------------
__global__ void gate_kernel(const float* __restrict__ x, const float* __restrict__ gw,
                            int* __restrict__ tok_e, float* __restrict__ tok_w,
                            int* __restrict__ counts) {
    int wave = threadIdx.x >> 5, lane = threadIdx.x & 31;
    int t = blockIdx.x * 8 + wave;
    if (t >= NTOK) return;
    const float* xr = x + (size_t)t * DIM;
    float acc[NE];
#pragma unroll
    for (int e = 0; e < NE; ++e) acc[e] = 0.0f;
    for (int d = lane; d < DIM; d += 32) {
        float xv = xr[d];
#pragma unroll
        for (int e = 0; e < NE; ++e) acc[e] += xv * gw[d * NE + e];
    }
#pragma unroll
    for (int e = 0; e < NE; ++e)
        for (int s = 16; s > 0; s >>= 1) acc[e] += __shfl_xor(acc[e], s, 32);
    if (lane == 0) {
        int e0 = 0; float l0 = acc[0];
        for (int e = 1; e < NE; ++e) if (acc[e] > l0) { l0 = acc[e]; e0 = e; }
        int e1 = (e0 == 0) ? 1 : 0; float l1 = acc[e1];
        for (int e = 0; e < NE; ++e)
            if (e != e0 && acc[e] > l1) { l1 = acc[e]; e1 = e; }
        float p1 = __expf(l1 - l0);                    // p0 = 1
        float inv = __builtin_amdgcn_rcpf(1.0f + p1);
        tok_e[t * 2 + 0] = e0; tok_e[t * 2 + 1] = e1;
        tok_w[t * 2 + 0] = inv; tok_w[t * 2 + 1] = p1 * inv;
        atomicAdd(&counts[e0], 1);
        atomicAdd(&counts[e1], 1);
    }
}

// ---------------- 128-aligned exclusive scan over 8 experts ----------------
__global__ void offsets_kernel(const int* __restrict__ counts, int* __restrict__ offs,
                               int* __restrict__ cursors) {
    if (threadIdx.x == 0) {
        int o = 0; offs[0] = 0;
        for (int e = 0; e < NE; ++e) {
            o += ((counts[e] + TILE_M - 1) / TILE_M) * TILE_M;
            offs[e + 1] = o;
        }
    }
    if (threadIdx.x < NE) cursors[threadIdx.x] = 0;
}

// ---------------- scatter tokens into expert-sorted slots ----------------
__global__ void scatter_kernel(const int* __restrict__ tok_e, const float* __restrict__ tok_w,
                               const int* __restrict__ offs, int* __restrict__ cursors,
                               int* __restrict__ slot_tok, float* __restrict__ slot_w) {
    int t = blockIdx.x * blockDim.x + threadIdx.x;
    if (t >= NTOK) return;
#pragma unroll
    for (int k = 0; k < NK; ++k) {
        int e = tok_e[t * 2 + k];
        int pos = offs[e] + atomicAdd(&cursors[e], 1);
        slot_tok[pos] = t;
        slot_w[pos] = tok_w[t * 2 + k];
    }
}

// Stage a 32(K) x 64(N) bf16 weight tile into LDS, K-transposed ([n][k]),
// using n-contiguous (k-strided) global d16 reads + one ds_store_b128 per thread.
__device__ __forceinline__ void stage_w_tile(const unsigned short* __restrict__ gsrc,
                                             size_t kstride,  // elements between k rows
                                             unsigned short (*sW)[TILE_K],
                                             int tid) {
    int n  = tid & 63;            // column handled by this thread
    int kb = (tid >> 6) * 8;      // 8 consecutive k values
    unsigned int pk[4];
    const unsigned short* p = gsrc + (size_t)kb * kstride + n;
#pragma unroll
    for (int j = 0; j < 4; ++j) {
        unsigned int lo = p[(2 * j + 0) * kstride];
        unsigned int hi = p[(2 * j + 1) * kstride];
        pk[j] = lo | (hi << 16);
    }
    *(uint4*)&sW[n][kb] = *(uint4*)pk;   // 16B-aligned: kb*2 bytes in {0,16,32,48}
}

// Load this wave's A fragment (16x32 bf16) straight from global:
// per ISA layout it is two lane-addressed 16B chunks at +half*16B and +32B.
__device__ __forceinline__ Frag16 load_a_frag(const unsigned short* __restrict__ arow,
                                              int k0, int half) {
    Frag16 a;
    uint4 a0 = *(const uint4*)(arow + k0 + half * 8);
    uint4 a1 = *(const uint4*)(arow + k0 + 16 + half * 8);
    a.u[0] = a0.x; a.u[1] = a0.y; a.u[2] = a0.z; a.u[3] = a0.w;
    a.u[4] = a1.x; a.u[5] = a1.y; a.u[6] = a1.z; a.u[7] = a1.w;
    return a;
}

// ---------------- FFN1: H = silu(X*W1) .* (X*W3), grouped by expert ----------------
__global__ __launch_bounds__(256) void ffn1_kernel(
        const unsigned short* __restrict__ xb,
        const unsigned short* __restrict__ w1b,
        const unsigned short* __restrict__ w3b,
        const int* __restrict__ offs,
        const int* __restrict__ slot_tok,
        unsigned short* __restrict__ H) {
    const int e = blockIdx.z;
    const int segBeg = offs[e], segLen = offs[e + 1] - segBeg;
    const int tileM = blockIdx.y * TILE_M;
    if (tileM >= segLen) return;                       // empty tile for this expert
    const int h0 = blockIdx.x * TILE_N;

    __shared__ unsigned short sW1[TILE_N][TILE_K];     // [n][k], K-pairs contiguous
    __shared__ unsigned short sW3[TILE_N][TILE_K];

    const int tid = threadIdx.x;
    const int lane = tid & 31, wave = tid >> 5;
    const int ln = lane & 15, half = lane >> 4;
    const int m0 = wave * 16;

    // each lane owns one A-row of its wave's strip: gather token id once
    const int rowTok = slot_tok[segBeg + tileM + m0 + ln];
    const unsigned short* arow = xb + (size_t)rowTok * DIM;

    const unsigned short* w1base = w1b + (size_t)e * DIM * HIDDEN + h0;
    const unsigned short* w3base = w3b + (size_t)e * DIM * HIDDEN + h0;

    f32x8 acc1[4], acc3[4];
#pragma unroll
    for (int c = 0; c < 4; ++c) { f32x8 z = {}; acc1[c] = z; acc3[c] = z; }

    for (int k0 = 0; k0 < DIM; k0 += TILE_K) {
        __syncthreads();
        stage_w_tile(w1base + (size_t)k0 * HIDDEN, HIDDEN, sW1, tid);
        stage_w_tile(w3base + (size_t)k0 * HIDDEN, HIDDEN, sW3, tid);
        // prefetch next K-step's weight tiles into cache (global_prefetch_b8)
        __builtin_prefetch(w1base + (size_t)(k0 + TILE_K) * HIDDEN + (tid & 63), 0, 0);
        __builtin_prefetch(w3base + (size_t)(k0 + TILE_K) * HIDDEN + (tid & 63), 0, 0);
        __syncthreads();

        Frag16 a = load_a_frag(arow, k0, half);        // global, no LDS for A
#pragma unroll
        for (int ct = 0; ct < 4; ++ct) {
            Frag16 b1, b3;
            const unsigned int* w1r = (const unsigned int*)&sW1[ct * 16 + ln][0];
            const unsigned int* w3r = (const unsigned int*)&sW3[ct * 16 + ln][0];
#pragma unroll
            for (int i = 0; i < 8; ++i) { b1.u[i] = w1r[pair_idx(i, half)]; b3.u[i] = w3r[pair_idx(i, half)]; }
            acc1[ct] = __builtin_amdgcn_wmma_f32_16x16x32_bf16(
                false, a.v, false, b1.v, (short)0, acc1[ct], false, false);
            acc3[ct] = __builtin_amdgcn_wmma_f32_16x16x32_bf16(
                false, a.v, false, b3.v, (short)0, acc3[ct], false, false);
        }
    }

    // epilogue: silu(a1)*a3 -> bf16 H, per C/D layout (VGPR j -> M = j + 8*half)
#pragma unroll
    for (int ct = 0; ct < 4; ++ct) {
#pragma unroll
        for (int j = 0; j < 8; ++j) {
            float v1 = acc1[ct][j], v3 = acc3[ct][j];
            float hh = v1 * fast_sigmoid(v1) * v3;
            int mrow = m0 + j + half * 8;
            int col = h0 + ct * 16 + ln;
            H[(size_t)(segBeg + tileM + mrow) * HIDDEN + col] = f32_to_bf16(hh);
        }
    }
}

// ---------------- FFN2: out[token] += w * (H * W2) ----------------
__global__ __launch_bounds__(256) void ffn2_kernel(
        const unsigned short* __restrict__ H,
        const unsigned short* __restrict__ w2b,
        const int* __restrict__ offs,
        const int* __restrict__ slot_tok,
        const float* __restrict__ slot_w,
        float* __restrict__ out) {
    const int e = blockIdx.z;
    const int segBeg = offs[e], segLen = offs[e + 1] - segBeg;
    const int tileM = blockIdx.y * TILE_M;
    if (tileM >= segLen) return;
    const int d0 = blockIdx.x * TILE_N;

    __shared__ unsigned short sW2[TILE_N][TILE_K];
    __shared__ int sTok[TILE_M];
    __shared__ float sWgt[TILE_M];

    const int tid = threadIdx.x;
    const int lane = tid & 31, wave = tid >> 5;
    const int ln = lane & 15, half = lane >> 4;
    const int m0 = wave * 16;

    if (tid < TILE_M) {
        sTok[tid] = slot_tok[segBeg + tileM + tid];
        sWgt[tid] = slot_w[segBeg + tileM + tid];
    }

    // lane's A row: contiguous slot rows of H
    const unsigned short* arow = H + (size_t)(segBeg + tileM + m0 + ln) * HIDDEN;
    const unsigned short* w2base = w2b + (size_t)e * HIDDEN * DIM + d0;

    f32x8 acc[4];
#pragma unroll
    for (int c = 0; c < 4; ++c) { f32x8 z = {}; acc[c] = z; }

    for (int k0 = 0; k0 < HIDDEN; k0 += TILE_K) {
        __syncthreads();
        stage_w_tile(w2base + (size_t)k0 * DIM, DIM, sW2, tid);
        __builtin_prefetch(w2base + (size_t)(k0 + TILE_K) * DIM + (tid & 63), 0, 0);
        __syncthreads();

        Frag16 a = load_a_frag(arow, k0, half);
#pragma unroll
        for (int ct = 0; ct < 4; ++ct) {
            Frag16 b2;
            const unsigned int* w2r = (const unsigned int*)&sW2[ct * 16 + ln][0];
#pragma unroll
            for (int i = 0; i < 8; ++i) b2.u[i] = w2r[pair_idx(i, half)];
            acc[ct] = __builtin_amdgcn_wmma_f32_16x16x32_bf16(
                false, a.v, false, b2.v, (short)0, acc[ct], false, false);
        }
    }

    // epilogue: scale by routing weight, atomic scatter-add per token
#pragma unroll
    for (int ct = 0; ct < 4; ++ct) {
#pragma unroll
        for (int j = 0; j < 8; ++j) {
            int mrow = m0 + j + half * 8;
            int col = d0 + ct * 16 + ln;
            float y = acc[ct][j] * sWgt[mrow];   // padding slots: weight == 0
            atomicAdd(&out[(size_t)sTok[mrow] * DIM + col], y);
        }
    }
}

// ---------------- host launcher ----------------
extern "C" void kernel_launch(void* const* d_in, const int* in_sizes, int n_in,
                              void* d_out, int out_size, void* d_ws, size_t ws_size,
                              hipStream_t stream) {
    const float* x  = (const float*)d_in[0];
    const float* gw = (const float*)d_in[1];
    const float* w1 = (const float*)d_in[2];
    const float* w3 = (const float*)d_in[3];
    const float* w2 = (const float*)d_in[4];
    float* out = (float*)d_out;
    (void)in_sizes; (void)n_in; (void)out_size; (void)ws_size;

    // carve workspace (~199 MB total)
    char* base = (char*)d_ws;
    size_t off = 0;
    auto carve = [&](size_t bytes) -> void* {
        off = (off + 255) & ~(size_t)255;
        void* p = base + off;
        off += bytes;
        return p;
    };
    unsigned short* xb  = (unsigned short*)carve((size_t)NTOK * DIM * 2);
    unsigned short* w1b = (unsigned short*)carve((size_t)NE * DIM * HIDDEN * 2);
    unsigned short* w3b = (unsigned short*)carve((size_t)NE * DIM * HIDDEN * 2);
    unsigned short* w2b = (unsigned short*)carve((size_t)NE * HIDDEN * DIM * 2);
    unsigned short* Hb  = (unsigned short*)carve((size_t)SLOT_CAP * HIDDEN * 2);
    int*   counts   = (int*)carve(NE * 4);
    int*   cursors  = (int*)carve(NE * 4);
    int*   offs     = (int*)carve((NE + 1) * 4);
    int*   tok_e    = (int*)carve((size_t)NTOK * 2 * 4);
    float* tok_w    = (float*)carve((size_t)NTOK * 2 * 4);
    int*   slot_tok = (int*)carve((size_t)SLOT_CAP * 4);
    float* slot_w   = (float*)carve((size_t)SLOT_CAP * 4);

    // 1) init output / routing state
    init_kernel<<<(NTOK * DIM + 255) / 256, 256, 0, stream>>>(out, slot_tok, slot_w, counts);

    // 2) fp32 -> bf16 (one-time per launch; GEMMs then stream bf16)
    cvt_bf16_kernel<<<(NTOK * DIM + 255) / 256, 256, 0, stream>>>(x, xb, NTOK * DIM);
    cvt_bf16_kernel<<<(NE * DIM * HIDDEN + 255) / 256, 256, 0, stream>>>(w1, w1b, NE * DIM * HIDDEN);
    cvt_bf16_kernel<<<(NE * DIM * HIDDEN + 255) / 256, 256, 0, stream>>>(w3, w3b, NE * DIM * HIDDEN);
    cvt_bf16_kernel<<<(NE * HIDDEN * DIM + 255) / 256, 256, 0, stream>>>(w2, w2b, NE * HIDDEN * DIM);

    // 3) gating (top-2 + softmax + counts)
    gate_kernel<<<NTOK / 8, 256, 0, stream>>>(x, gw, tok_e, tok_w, counts);

    // 4) aligned segment offsets + scatter into expert-sorted slots
    offsets_kernel<<<1, 32, 0, stream>>>(counts, offs, cursors);
    scatter_kernel<<<(NTOK + 255) / 256, 256, 0, stream>>>(tok_e, tok_w, offs, cursors,
                                                           slot_tok, slot_w);

    // 5) grouped GEMM #1+#2 fused: H = silu(X W1) * (X W3)
    dim3 g1(HIDDEN / TILE_N, NTOK / TILE_M, NE);   // 44 x 32 x 8, empty tiles early-exit
    ffn1_kernel<<<g1, 256, 0, stream>>>(xb, w1b, w3b, offs, slot_tok, Hb);

    // 6) grouped GEMM #3 + weighted scatter-add
    dim3 g2(DIM / TILE_N, NTOK / TILE_M, NE);      // 16 x 32 x 8
    ffn2_kernel<<<g2, 256, 0, stream>>>(Hb, w2b, offs, slot_tok, slot_w, out);
}
